// GAE_1185410974360
// MI455X (gfx1250) — matrix operand order repaired and checked by reading, not compiled, not run on hardware
//
#include <hip/hip_runtime.h>

typedef __attribute__((ext_vector_type(16))) __bf16 v16bf;
typedef __attribute__((ext_vector_type(8)))  float  v8f;

#define NN_    65536
#define NGRAPH 32
#define NSLOT  2048
#define EDGES  1048576
#define EU     65536
#define DIN    64
#define HID    128
#define LAT    32

// ---- WMMA operand loaders (bf16 from f32 row-major) -------------------------
// A: 16x32 tile, rows row0..row0+15, K-block kb..kb+31.
// Contiguous k order used here (same as loadB): lane<16 holds K {kb..kb+15}; lane>=16 holds {kb+16..kb+31} (elem e -> K=kb+16*kh+e).
__device__ __forceinline__ v16bf loadA_bf16(const float* __restrict__ S, int ld,
                                            int row0, int kb, float rscale) {
  const int lane = threadIdx.x & 31;
  const int r = lane & 15, kh = lane >> 4;
  const float* p = S + (size_t)(row0 + r) * ld + kb + kh * 16;
  v16bf a;
#pragma unroll
  for (int e = 0; e < 8; ++e) a[e]     = (__bf16)(p[e]      * rscale);
#pragma unroll
  for (int e = 0; e < 8; ++e) a[8 + e] = (__bf16)(p[8 + e] * rscale);
  return a;
}

// B: 32x16 tile of W (fi x fo row-major), K rows kb..kb+31, cols col0..col0+15.
// lanes 0-15: K=kb..kb+15 (elem e -> K=kb+e); lanes 16-31: K=kb+16..kb+31.
__device__ __forceinline__ v16bf loadB_bf16(const float* __restrict__ W, int ld,
                                            int kb, int col0) {
  const int lane = threadIdx.x & 31;
  const int n = lane & 15, kh = lane >> 4;
  const float* p = W + (size_t)(kb + kh * 16) * ld + col0 + n;
  v16bf b;
#pragma unroll
  for (int e = 0; e < 16; ++e) b[e] = (__bf16)(p[(size_t)e * ld]);
  return b;
}

// ---- degree + scatter-add (segment mean numerator) --------------------------
__global__ void k_degree(const int* __restrict__ dst, float* __restrict__ deg) {
  int e = blockIdx.x * blockDim.x + threadIdx.x;
  if (e < EDGES) atomicAdd(&deg[dst[e]], 1.0f);
}

template <int FI>
__global__ void k_scatter(const float* __restrict__ feat, const int* __restrict__ src,
                          const int* __restrict__ dst, float* __restrict__ agg) {
  const int cpe = FI / 4;
  long long t = (long long)blockIdx.x * blockDim.x + threadIdx.x;
  int e = (int)(t / cpe);
  if (e >= EDGES) return;
  int c = (int)(t % cpe) * 4;
  int s = src[e], d = dst[e];
  float4 v = *(const float4*)(feat + (size_t)s * FI + c);
  float* pd = agg + (size_t)d * FI + c;
  atomicAdd(pd + 0, v.x); atomicAdd(pd + 1, v.y);
  atomicAdd(pd + 2, v.z); atomicAdd(pd + 3, v.w);
}

// half-wave (16-lane) sum reduction: 4 shuffle steps on wave32
__device__ __forceinline__ float hw16_sum(float v) {
  v += __shfl_xor(v, 8, 16);
  v += __shfl_xor(v, 4, 16);
  v += __shfl_xor(v, 2, 16);
  v += __shfl_xor(v, 1, 16);
  return v;
}

// ---- fused SAGE layer: out = act(LN(mean@Wl + b + x@Wr)) --------------------
// One wave per 16x16 output tile; FO/16 waves per block; 16 rows per block.
template <int FI, int FO, bool LNRELU, bool CLAMP>
__global__ void k_sage(const float* __restrict__ agg, const float* __restrict__ xin,
                       const float* __restrict__ deg,
                       const float* __restrict__ Wl, const float* __restrict__ Wr,
                       const float* __restrict__ bias,
                       const float* __restrict__ gamma, const float* __restrict__ beta,
                       float* __restrict__ out) {
  const int wave = threadIdx.x >> 5;
  const int col0 = wave * 16;
  const int row0 = blockIdx.x * 16;
  const int lane = threadIdx.x & 31;
  const int r = lane & 15, half = lane >> 4, n = lane & 15;

  const float inv = 1.0f / fmaxf(deg[row0 + r], 1.0f);

  v8f c = {0.f, 0.f, 0.f, 0.f, 0.f, 0.f, 0.f, 0.f};
#pragma unroll
  for (int kb = 0; kb < FI; kb += 32) {   // mean @ Wl
    v16bf a = loadA_bf16(agg, FI, row0, kb, inv);
    v16bf b = loadB_bf16(Wl, FO, kb, col0);
    c = __builtin_amdgcn_wmma_f32_16x16x32_bf16(false, a, false, b, (short)0, c, false, false);
  }
#pragma unroll
  for (int kb = 0; kb < FI; kb += 32) {   // x @ Wr (accumulate)
    v16bf a = loadA_bf16(xin, FI, row0, kb, 1.0f);
    v16bf b = loadB_bf16(Wr, FO, kb, col0);
    c = __builtin_amdgcn_wmma_f32_16x16x32_bf16(false, a, false, b, (short)0, c, false, false);
  }
  const float bc = bias[col0 + n];

  if (!LNRELU) {
    // direct register -> global store (C/D layout: row = i + 8*half, col = n)
#pragma unroll
    for (int i = 0; i < 8; ++i) {
      float v = c[i] + bc;
      if (CLAMP) v = fminf(v, 10.0f);
      out[(size_t)(row0 + i + 8 * half) * FO + col0 + n] = v;
    }
    return;
  }

  // ---- LayerNorm + ReLU epilogue (FO == 128 here) ----
  __shared__ float tile[16 * FO];
#pragma unroll
  for (int i = 0; i < 8; ++i)
    tile[(i + 8 * half) * FO + col0 + n] = c[i] + bc;
  __syncthreads();

  constexpr int EL = FO / 16;               // elements per thread (8)
  const int row = threadIdx.x >> 4;         // 0..15 (blockDim == 256)
  const int sub = threadIdx.x & 15;         // contiguous half-wave per row
  float vals[EL];
  float s = 0.f;
#pragma unroll
  for (int k = 0; k < EL; ++k) { vals[k] = tile[row * FO + sub + 16 * k]; s += vals[k]; }
  const float mean = hw16_sum(s) * (1.0f / (float)FO);
  float vs = 0.f;
#pragma unroll
  for (int k = 0; k < EL; ++k) { float d = vals[k] - mean; vs += d * d; }
  const float rstd = __frsqrt_rn(hw16_sum(vs) * (1.0f / (float)FO) + 1e-5f);
#pragma unroll
  for (int k = 0; k < EL; ++k) {
    const int j = sub + 16 * k;
    float h = (vals[k] - mean) * rstd * gamma[j] + beta[j];
    out[(size_t)(row0 + row) * FO + j] = fmaxf(h, 0.f);
  }
}

// ---- Z scatter (honors node_slot) ------------------------------------------
__global__ void k_zscatter(const float* __restrict__ mu, const int* __restrict__ slot,
                           float* __restrict__ Z) {
  int t = blockIdx.x * blockDim.x + threadIdx.x;   // NN*8 threads
  int i = t >> 3, c = (t & 7) * 4;
  int s = slot[i];
  *(float4*)(Z + (size_t)s * LAT + c) = *(const float4*)(mu + (size_t)i * LAT + c);
}

// ---- edge logits: scale * <Z[b,us], Z[b,ud]> + bias -------------------------
__global__ void k_edge(const float* __restrict__ Z, const int* __restrict__ us,
                       const int* __restrict__ ud, const float* __restrict__ scale,
                       const float* __restrict__ bias, float* __restrict__ out) {
  int idx = blockIdx.x * blockDim.x + threadIdx.x;   // B*EU threads
  int b = idx >> 16, e = idx & (EU - 1);
  const float4* pa = (const float4*)(Z + ((size_t)b * NSLOT + us[e]) * LAT);
  const float4* pb = (const float4*)(Z + ((size_t)b * NSLOT + ud[e]) * LAT);
  float dot = 0.f;
#pragma unroll
  for (int i = 0; i < 8; ++i) {
    float4 A = pa[i], B = pb[i];
    dot += A.x * B.x + A.y * B.y + A.z * B.z + A.w * B.w;
  }
  out[idx] = scale[0] * dot + bias[0];
}

// ---- ignition head: (relu(Z@iW1 + ib1) @ iW2 + ib2) -------------------------
__global__ void k_ign(const float* __restrict__ Z, const float* __restrict__ iW1,
                      const float* __restrict__ ib1, const float* __restrict__ iW2,
                      const float* __restrict__ ib2, float* __restrict__ out) {
  __shared__ float tile[16 * HID];
  const int wave = threadIdx.x >> 5;
  const int col0 = wave * 16;
  const int row0 = blockIdx.x * 16;
  const int lane = threadIdx.x & 31;
  const int half = lane >> 4, n = lane & 15;

  v8f c = {0.f, 0.f, 0.f, 0.f, 0.f, 0.f, 0.f, 0.f};
  v16bf a = loadA_bf16(Z, LAT, row0, 0, 1.0f);        // K = 32 = LAT, one step
  v16bf b = loadB_bf16(iW1, HID, 0, col0);
  c = __builtin_amdgcn_wmma_f32_16x16x32_bf16(false, a, false, b, (short)0, c, false, false);

  const float w2 = iW2[col0 + n];
  const float b1 = ib1[col0 + n];
#pragma unroll
  for (int i = 0; i < 8; ++i)
    tile[(i + 8 * half) * HID + col0 + n] = fmaxf(c[i] + b1, 0.f) * w2;
  __syncthreads();

  const int row = threadIdx.x >> 4;        // 16 threads per row, half-wave reduce
  const int sub = threadIdx.x & 15;
  float s = 0.f;
#pragma unroll
  for (int k = 0; k < HID / 16; ++k) s += tile[row * HID + sub + 16 * k];
  s = hw16_sum(s);
  if (sub == 0) out[row0 + row] = s + ib2[0];
}

// ---- driver -----------------------------------------------------------------
extern "C" void kernel_launch(void* const* d_in, const int* in_sizes, int n_in,
                              void* d_out, int out_size, void* d_ws, size_t ws_size,
                              hipStream_t stream) {
  (void)in_sizes; (void)n_in; (void)out_size; (void)ws_size;
  const float* x  = (const float*)d_in[0];
  const int* ei   = (const int*)d_in[1];
  const int* src  = ei;
  const int* dst  = ei + EDGES;
  const int* slot = (const int*)d_in[2];
  const int* usrc = (const int*)d_in[3];
  const int* udst = (const int*)d_in[4];
  // params 5..36: per tower (13): W0l,W0r,b0, W1l,W1r,b1, W2l,W2r,b2, g0,bt0, g1,bt1
  const float* iW1 = (const float*)d_in[31];
  const float* ib1 = (const float*)d_in[32];
  const float* iW2 = (const float*)d_in[33];
  const float* ib2 = (const float*)d_in[34];
  const float* esc = (const float*)d_in[35];
  const float* ebi = (const float*)d_in[36];

  // workspace layout
  char* ws = (char*)d_ws;
  float* agg  = (float*)(ws + 0);                 // NN*HID f32 = 32 MiB (later reused as Z)
  float* hA   = (float*)(ws + 33554432);          // NN*HID
  float* hB   = (float*)(ws + 67108864);          // NN*HID
  float* deg  = (float*)(ws + 100663296);         // NN f32
  float* aggx = (float*)(ws + 100925440);         // NN*DIN f32 (shared layer-0 aggregation)
  float* Z    = agg;                              // NN*LAT, reused after towers done

  // output layout: edge_u_logits | ign_logits | mu | logstd
  float* out_edge   = (float*)d_out;
  float* out_ign    = out_edge + (size_t)NGRAPH * EU;
  float* out_mu     = out_ign + NN_;
  float* out_logstd = out_mu + (size_t)NN_ * LAT;

  // degree (shared across all layers) + shared layer-0 aggregation of x
  hipMemsetAsync(deg, 0, (size_t)NN_ * 4, stream);
  k_degree<<<EDGES / 256, 256, 0, stream>>>(dst, deg);
  hipMemsetAsync(aggx, 0, (size_t)NN_ * DIN * 4, stream);
  k_scatter<DIN><<<EDGES * (DIN / 4) / 256, 256, 0, stream>>>(x, src, dst, aggx);

  for (int t = 0; t < 2; ++t) {
    const int base = 5 + 13 * t;
    auto g = [&](int i) { return (const float*)d_in[base + i]; };
    // layer 0: 64 -> 128, LN+ReLU
    k_sage<DIN, HID, true, false><<<NN_ / 16, (HID / 16) * 32, 0, stream>>>(
        aggx, x, deg, g(0), g(1), g(2), g(9), g(10), hA);
    // layer 1: 128 -> 128, LN+ReLU
    hipMemsetAsync(agg, 0, (size_t)NN_ * HID * 4, stream);
    k_scatter<HID><<<EDGES * (HID / 4) / 256, 256, 0, stream>>>(hA, src, dst, agg);
    k_sage<HID, HID, true, false><<<NN_ / 16, (HID / 16) * 32, 0, stream>>>(
        agg, hA, deg, g(3), g(4), g(5), g(11), g(12), hB);
    // layer 2: 128 -> 32, no norm/act; logstd clamped at 10
    hipMemsetAsync(agg, 0, (size_t)NN_ * HID * 4, stream);
    k_scatter<HID><<<EDGES * (HID / 4) / 256, 256, 0, stream>>>(hB, src, dst, agg);
    if (t == 0)
      k_sage<HID, LAT, false, false><<<NN_ / 16, (LAT / 16) * 32, 0, stream>>>(
          agg, hB, deg, g(6), g(7), g(8), nullptr, nullptr, out_mu);
    else
      k_sage<HID, LAT, false, true><<<NN_ / 16, (LAT / 16) * 32, 0, stream>>>(
          agg, hB, deg, g(6), g(7), g(8), nullptr, nullptr, out_logstd);
  }

  // decode heads (agg buffer is free now -> Z)
  hipMemsetAsync(Z, 0, (size_t)NN_ * LAT * 4, stream);
  k_zscatter<<<NN_ * (LAT / 4) / 256, 256, 0, stream>>>(out_mu, slot, Z);
  k_edge<<<(NGRAPH * EU) / 256, 256, 0, stream>>>(Z, usrc, udst, esc, ebi, out_edge);
  k_ign<<<NN_ / 16, (HID / 16) * 32, 0, stream>>>(Z, iW1, ib1, iW2, ib2, out_ign);
}
